// Decoder_GRU_33363305955818
// MI455X (gfx1250) — matrix operand order repaired and checked by reading, not compiled, not run on hardware
//
#include <hip/hip_runtime.h>

#define NN 2048
#define BB 16
#define FF 64
#define F2 128
#define TT 8
#define MROWS (BB * NN) /* 32768 */

typedef __bf16 bf16;
typedef __bf16 v16bf __attribute__((ext_vector_type(16)));
typedef __bf16 v8bf  __attribute__((ext_vector_type(8)));
typedef float  v8f   __attribute__((ext_vector_type(8)));
typedef int    v4i   __attribute__((ext_vector_type(4)));

typedef __attribute__((address_space(1))) v4i gv4i;   // global int4
typedef __attribute__((address_space(3))) v4i lv4i;   // LDS int4

union AccU  { v8f v; float f[8]; };
union FragU { v16bf v; v8bf h[2]; };

#if defined(__has_builtin)
#if __has_builtin(__builtin_amdgcn_global_load_async_to_lds_b128) && \
    __has_builtin(__builtin_amdgcn_s_wait_asynccnt)
#define HAVE_ASYNC_LDS 1
#endif
#endif
#ifndef HAVE_ASYNC_LDS
#define HAVE_ASYNC_LDS 0
#endif

__device__ __forceinline__ float sigf(float x) { return 1.0f / (1.0f + __expf(-x)); }

// ---------------- f32 -> bf16 (same layout) ----------------
__global__ void k_cvt_bf16(const float* __restrict__ s, bf16* __restrict__ d, int n) {
  for (int i = blockIdx.x * blockDim.x + threadIdx.x; i < n; i += gridDim.x * blockDim.x)
    d[i] = (bf16)s[i];
}
// ---------------- f32 [R][C] -> bf16 [C][R] (weights, one-time) ----------------
__global__ void k_transposeT(const float* __restrict__ s, bf16* __restrict__ d, int R, int C) {
  int total = R * C;
  for (int i = blockIdx.x * blockDim.x + threadIdx.x; i < total; i += gridDim.x * blockDim.x) {
    int r = i / C, c = i - r * C;
    d[(size_t)c * R + r] = (bf16)s[i];
  }
}
// ---------------- x [B][N][F] -> xT [B][F][N] (one-time) ----------------
__global__ void k_transpose_x(const float* __restrict__ s, bf16* __restrict__ d) {
  int total = BB * NN * FF;
  for (int i = blockIdx.x * blockDim.x + threadIdx.x; i < total; i += gridDim.x * blockDim.x) {
    int b = i / (NN * FF);
    int rem = i - b * NN * FF;
    int n = rem / FF, f = rem - n * FF;
    d[((size_t)b * FF + f) * NN + n] = (bf16)s[i];
  }
}

// ---------------- LC = L @ concat(x, h) per batch ----------------
// A = L (row-major, K contiguous). B = xT/hT (feature-major, K contiguous).
// grid: (NN/128, BB), block 256 (8 waves). Out tile 128x128.
#if HAVE_ASYNC_LDS
// Double-buffered async global->LDS pipeline, BK=32, ASYNCcnt-tracked.
__global__ __launch_bounds__(256) void k_gemm_lc(
    const bf16* __restrict__ Lb, const bf16* __restrict__ xT,
    const bf16* __restrict__ hT, bf16* __restrict__ LC)
{
  __shared__ bf16 As[2][128][40];
  __shared__ bf16 Bs[2][128][40];
  const int tid = threadIdx.x;
  const int lane = tid & 31;
  const int w  = tid >> 5;
  const int wm = w & 3, wn = w >> 2;
  const int ln = lane & 15, lh = lane >> 4;
  const int m0 = blockIdx.x * 128;
  const int b  = blockIdx.y;
  const bf16* xb = xT + (size_t)b * FF * NN;
  const bf16* hb = hT + (size_t)b * FF * NN;

  AccU acc[2][4];
#pragma unroll
  for (int i = 0; i < 2; ++i)
#pragma unroll
    for (int j = 0; j < 4; ++j)
#pragma unroll
      for (int r = 0; r < 8; ++r) acc[i][j].f[r] = 0.0f;

  auto issue = [&](int k0, int buf) {
#pragma unroll
    for (int i = 0; i < 2; ++i) {           // A tile: 128x32
      int id = tid + i * 256, rr = id >> 2, c = (id & 3) << 3;
      __builtin_amdgcn_global_load_async_to_lds_b128(
          (gv4i*)(Lb + (size_t)(m0 + rr) * NN + k0 + c),
          (lv4i*)&As[buf][rr][c], 0, 0);
    }
#pragma unroll
    for (int i = 0; i < 2; ++i) {           // B tile: 128 cols x 32 k
      int id = tid + i * 256, rr = id >> 2, c = (id & 3) << 3;
      const bf16* src = (rr < FF) ? (xb + (size_t)rr * NN) : (hb + (size_t)(rr - FF) * NN);
      __builtin_amdgcn_global_load_async_to_lds_b128(
          (gv4i*)(src + k0 + c),
          (lv4i*)&Bs[buf][rr][c], 0, 0);
    }
  };

  issue(0, 0);
  for (int k0 = 0; k0 < NN; k0 += 32) {
    const int buf = (k0 >> 5) & 1;
    if (k0 + 32 < NN) {
      issue(k0 + 32, buf ^ 1);
      __builtin_amdgcn_s_wait_asynccnt(4);   // current chunk landed (in-order)
    } else {
      __builtin_amdgcn_s_wait_asynccnt(0);
    }
    __syncthreads();
    FragU a[2], bq[4];
#pragma unroll
    for (int mi = 0; mi < 2; ++mi) {
      int row = wm * 32 + mi * 16 + ln;
      int kb  = lh * 8;
      a[mi].h[0] = *(const v8bf*)&As[buf][row][kb];
      a[mi].h[1] = *(const v8bf*)&As[buf][row][kb + 16];
    }
#pragma unroll
    for (int ni = 0; ni < 4; ++ni) {
      int col = wn * 64 + ni * 16 + ln;
      int kb  = lh * 8;
      bq[ni].h[0] = *(const v8bf*)&Bs[buf][col][kb];
      bq[ni].h[1] = *(const v8bf*)&Bs[buf][col][kb + 16];
    }
#pragma unroll
    for (int mi = 0; mi < 2; ++mi)
#pragma unroll
      for (int ni = 0; ni < 4; ++ni)
        acc[mi][ni].v = __builtin_amdgcn_wmma_f32_16x16x32_bf16(
            false, a[mi].v, false, bq[ni].v, (short)0, acc[mi][ni].v, false, false);
    __syncthreads();
  }
#pragma unroll
  for (int mi = 0; mi < 2; ++mi)
#pragma unroll
    for (int ni = 0; ni < 4; ++ni) {
      int gm = m0 + wm * 32 + mi * 16 + lh * 8;
      int gn = wn * 64 + ni * 16 + ln;
      bf16* dst = LC + ((size_t)b * NN + gm) * F2 + gn;
#pragma unroll
      for (int r = 0; r < 8; ++r) dst[(size_t)r * F2] = (bf16)acc[mi][ni].f[r];
    }
}
#else
// Fallback: BK=64 single-buffer with register-stage prefetch.
__global__ __launch_bounds__(256) void k_gemm_lc(
    const bf16* __restrict__ Lb, const bf16* __restrict__ xT,
    const bf16* __restrict__ hT, bf16* __restrict__ LC)
{
  __shared__ bf16 As[128][72];
  __shared__ bf16 Bs[128][72];
  const int tid = threadIdx.x;
  const int lane = tid & 31;
  const int w  = tid >> 5;
  const int wm = w & 3, wn = w >> 2;
  const int ln = lane & 15, lh = lane >> 4;
  const int m0 = blockIdx.x * 128;
  const int b  = blockIdx.y;
  const bf16* xb = xT + (size_t)b * FF * NN;
  const bf16* hb = hT + (size_t)b * FF * NN;

  AccU acc[2][4];
#pragma unroll
  for (int i = 0; i < 2; ++i)
#pragma unroll
    for (int j = 0; j < 4; ++j)
#pragma unroll
      for (int r = 0; r < 8; ++r) acc[i][j].f[r] = 0.0f;

  v8bf regA[4], regB[4];
#pragma unroll
  for (int i = 0; i < 4; ++i) {
    int id = tid + i * 256, rr = id >> 3, c = (id & 7) << 3;
    regA[i] = *(const v8bf*)(Lb + (size_t)(m0 + rr) * NN + c);
    const bf16* src = (rr < FF) ? (xb + (size_t)rr * NN) : (hb + (size_t)(rr - FF) * NN);
    regB[i] = *(const v8bf*)(src + c);
  }

  for (int k0 = 0; k0 < NN; k0 += 64) {
#pragma unroll
    for (int i = 0; i < 4; ++i) {
      int id = tid + i * 256, rr = id >> 3, c = (id & 7) << 3;
      *(v8bf*)&As[rr][c] = regA[i];
      *(v8bf*)&Bs[rr][c] = regB[i];
    }
    __syncthreads();
    if (k0 + 64 < NN) {
      int k1 = k0 + 64;
#pragma unroll
      for (int i = 0; i < 4; ++i) {
        int id = tid + i * 256, rr = id >> 3, c = (id & 7) << 3;
        regA[i] = *(const v8bf*)(Lb + (size_t)(m0 + rr) * NN + k1 + c);
        const bf16* src = (rr < FF) ? (xb + (size_t)rr * NN) : (hb + (size_t)(rr - FF) * NN);
        regB[i] = *(const v8bf*)(src + k1 + c);
      }
    }
#pragma unroll
    for (int kk = 0; kk < 64; kk += 32) {
      FragU a[2], bq[4];
#pragma unroll
      for (int mi = 0; mi < 2; ++mi) {
        int row = wm * 32 + mi * 16 + ln;
        int kb  = kk + lh * 8;
        a[mi].h[0] = *(const v8bf*)&As[row][kb];
        a[mi].h[1] = *(const v8bf*)&As[row][kb + 16];
      }
#pragma unroll
      for (int ni = 0; ni < 4; ++ni) {
        int col = wn * 64 + ni * 16 + ln;
        int kb  = kk + lh * 8;
        bq[ni].h[0] = *(const v8bf*)&Bs[col][kb];
        bq[ni].h[1] = *(const v8bf*)&Bs[col][kb + 16];
      }
#pragma unroll
      for (int mi = 0; mi < 2; ++mi)
#pragma unroll
        for (int ni = 0; ni < 4; ++ni)
          acc[mi][ni].v = __builtin_amdgcn_wmma_f32_16x16x32_bf16(
              false, a[mi].v, false, bq[ni].v, (short)0, acc[mi][ni].v, false, false);
    }
    __syncthreads();
  }
#pragma unroll
  for (int mi = 0; mi < 2; ++mi)
#pragma unroll
    for (int ni = 0; ni < 4; ++ni) {
      int gm = m0 + wm * 32 + mi * 16 + lh * 8;
      int gn = wn * 64 + ni * 16 + ln;
      bf16* dst = LC + ((size_t)b * NN + gm) * F2 + gn;
#pragma unroll
      for (int r = 0; r < 8; ++r) dst[(size_t)r * F2] = (bf16)acc[mi][ni].f[r];
    }
}
#endif

// -------- gates = [x|h|LC] @ Wg + bg ; fused r,u epilogue --------
__global__ __launch_bounds__(256) void k_gates(
    const bf16* __restrict__ xbf, const bf16* __restrict__ hbf,
    const bf16* __restrict__ LC,  const bf16* __restrict__ WgT,
    const float* __restrict__ bg, const float* __restrict__ hxT,
    bf16* __restrict__ rh, bf16* __restrict__ rhT, float* __restrict__ uT)
{
  __shared__ bf16 As[128][72];
  __shared__ bf16 Bs[128][72];
  const int tid = threadIdx.x;
  const int lane = tid & 31;
  const int w  = tid >> 5;
  const int wm = w & 3, wn = w >> 2;
  const int ln = lane & 15, lh = lane >> 4;
  const int p0 = blockIdx.x * 128;

  AccU acc[2][4];
#pragma unroll
  for (int i = 0; i < 2; ++i)
#pragma unroll
    for (int j = 0; j < 4; ++j)
#pragma unroll
      for (int r = 0; r < 8; ++r) acc[i][j].f[r] = 0.0f;

  for (int k0 = 0; k0 < 256; k0 += 64) {
    const bf16* abase; int astride, aoff;
    if (k0 == 0)        { abase = xbf; astride = FF; aoff = 0;  }
    else if (k0 == 64)  { abase = hbf; astride = FF; aoff = 0;  }
    else if (k0 == 128) { abase = LC;  astride = F2; aoff = 0;  }
    else                { abase = LC;  astride = F2; aoff = FF; }
#pragma unroll
    for (int i = 0; i < 4; ++i) {
      int id = tid + i * 256, rr = id >> 3, c = (id & 7) << 3;
      *(v8bf*)&As[rr][c] = *(const v8bf*)(abase + (size_t)(p0 + rr) * astride + aoff + c);
      *(v8bf*)&Bs[rr][c] = *(const v8bf*)(WgT + (size_t)rr * 256 + k0 + c);
    }
    __syncthreads();
#pragma unroll
    for (int kk = 0; kk < 64; kk += 32) {
      FragU a[2], bq[4];
#pragma unroll
      for (int mi = 0; mi < 2; ++mi) {
        int row = wm * 32 + mi * 16 + ln;
        int kb  = kk + lh * 8;
        a[mi].h[0] = *(const v8bf*)&As[row][kb];
        a[mi].h[1] = *(const v8bf*)&As[row][kb + 16];
      }
#pragma unroll
      for (int ni = 0; ni < 4; ++ni) {
        int col = wn * 64 + ni * 16 + ln;
        int kb  = kk + lh * 8;
        bq[ni].h[0] = *(const v8bf*)&Bs[col][kb];
        bq[ni].h[1] = *(const v8bf*)&Bs[col][kb + 16];
      }
#pragma unroll
      for (int mi = 0; mi < 2; ++mi)
#pragma unroll
        for (int ni = 0; ni < 4; ++ni)
          acc[mi][ni].v = __builtin_amdgcn_wmma_f32_16x16x32_bf16(
              false, a[mi].v, false, bq[ni].v, (short)0, acc[mi][ni].v, false, false);
    }
    __syncthreads();
  }
  const int b  = p0 >> 11;
#pragma unroll
  for (int mi = 0; mi < 2; ++mi)
#pragma unroll
    for (int ni = 0; ni < 4; ++ni) {
      int gm = p0 + wm * 32 + mi * 16 + lh * 8;
      int gn = wn * 64 + ni * 16 + ln;
      int n0 = gm - (b << 11);
      if (gn < FF) {
        AccU hv; hv.v = *(const v8f*)(hxT + (size_t)gn * MROWS + gm);
        v8bf pk;
#pragma unroll
        for (int r = 0; r < 8; ++r) {
          float rv = sigf(acc[mi][ni].f[r] + bg[gn]) * hv.f[r];
          rh[(size_t)(gm + r) * FF + gn] = (bf16)rv;
          pk[r] = (bf16)rv;
        }
        *(v8bf*)(rhT + ((size_t)(b * FF + gn)) * NN + n0) = pk;
      } else {
        AccU t;
#pragma unroll
        for (int r = 0; r < 8; ++r) t.f[r] = sigf(acc[mi][ni].f[r] + bg[gn]);
        *(v8f*)(uT + (size_t)(gn - FF) * MROWS + gm) = t.v;
      }
    }
}

// ---- cy = tanh([x|rh|LC] @ Wu + bu); hy = u*h + (1-u)*cy ----
__global__ __launch_bounds__(256) void k_update(
    const bf16* __restrict__ xbf, const bf16* __restrict__ rh,
    const bf16* __restrict__ LC,  const bf16* __restrict__ WuT,
    const float* __restrict__ bu, const float* __restrict__ uT,
    float* __restrict__ hxT, bf16* __restrict__ hT, bf16* __restrict__ hbf)
{
  __shared__ bf16 As[128][72];
  __shared__ bf16 Bs[64][72];
  const int tid = threadIdx.x;
  const int lane = tid & 31;
  const int w  = tid >> 5;
  const int ln = lane & 15, lh = lane >> 4;
  const int p0 = blockIdx.x * 128;

  AccU acc[4];
#pragma unroll
  for (int j = 0; j < 4; ++j)
#pragma unroll
    for (int r = 0; r < 8; ++r) acc[j].f[r] = 0.0f;

  for (int k0 = 0; k0 < 256; k0 += 64) {
    const bf16* abase; int astride, aoff;
    if (k0 == 0)        { abase = xbf; astride = FF; aoff = 0;  }
    else if (k0 == 64)  { abase = rh;  astride = FF; aoff = 0;  }
    else if (k0 == 128) { abase = LC;  astride = F2; aoff = 0;  }
    else                { abase = LC;  astride = F2; aoff = FF; }
#pragma unroll
    for (int i = 0; i < 4; ++i) {
      int id = tid + i * 256, rr = id >> 3, c = (id & 7) << 3;
      *(v8bf*)&As[rr][c] = *(const v8bf*)(abase + (size_t)(p0 + rr) * astride + aoff + c);
    }
#pragma unroll
    for (int i = 0; i < 2; ++i) {
      int id = tid + i * 256, col = id >> 3, c = (id & 7) << 3;
      *(v8bf*)&Bs[col][c] = *(const v8bf*)(WuT + (size_t)col * 256 + k0 + c);
    }
    __syncthreads();
#pragma unroll
    for (int kk = 0; kk < 64; kk += 32) {
      FragU a, bq[4];
      {
        int row = w * 16 + ln;
        int kb  = kk + lh * 8;
        a.h[0] = *(const v8bf*)&As[row][kb];
        a.h[1] = *(const v8bf*)&As[row][kb + 16];
      }
#pragma unroll
      for (int ni = 0; ni < 4; ++ni) {
        int col = ni * 16 + ln;
        int kb  = kk + lh * 8;
        bq[ni].h[0] = *(const v8bf*)&Bs[col][kb];
        bq[ni].h[1] = *(const v8bf*)&Bs[col][kb + 16];
      }
#pragma unroll
      for (int ni = 0; ni < 4; ++ni)
        acc[ni].v = __builtin_amdgcn_wmma_f32_16x16x32_bf16(
            false, a.v, false, bq[ni].v, (short)0, acc[ni].v, false, false);
    }
    __syncthreads();
  }
  const int b = p0 >> 11;
#pragma unroll
  for (int ni = 0; ni < 4; ++ni) {
    int col = ni * 16 + ln;
    int gm  = p0 + w * 16 + lh * 8;
    int n0  = gm - (b << 11);
    AccU uu; uu.v = *(const v8f*)(uT  + (size_t)col * MROWS + gm);
    AccU hv; hv.v = *(const v8f*)(hxT + (size_t)col * MROWS + gm);
    AccU hy; v8bf pk;
#pragma unroll
    for (int r = 0; r < 8; ++r) {
      float cy = tanhf(acc[ni].f[r] + bu[col]);
      float h2 = uu.f[r] * hv.f[r] + (1.0f - uu.f[r]) * cy;
      hy.f[r] = h2;
      pk[r]   = (bf16)h2;
      hbf[(size_t)(gm + r) * FF + col] = (bf16)h2;
    }
    *(v8f*)(hxT + (size_t)col * MROWS + gm) = hy.v;
    *(v8bf*)(hT + ((size_t)(b * FF + col)) * NN + n0) = pk;
  }
}

// -------- yt^T GEMM: D[f][p] = WeT[f][k] * h[p][k] ; sigmoid epilogue --------
__global__ __launch_bounds__(256) void k_edge(
    const bf16* __restrict__ hbf, const bf16* __restrict__ WeT,
    const float* __restrict__ be, float* __restrict__ outp)
{
  __shared__ bf16 As[64][72];
  __shared__ bf16 Bs[128][72];
  const int tid = threadIdx.x;
  const int lane = tid & 31;
  const int w  = tid >> 5;
  const int ln = lane & 15, lh = lane >> 4;
  const int p0 = blockIdx.x * 128;

  AccU acc[4];
#pragma unroll
  for (int j = 0; j < 4; ++j)
#pragma unroll
    for (int r = 0; r < 8; ++r) acc[j].f[r] = 0.0f;

#pragma unroll
  for (int i = 0; i < 2; ++i) {
    int id = tid + i * 256, f = id >> 3, c = (id & 7) << 3;
    *(v8bf*)&As[f][c] = *(const v8bf*)(WeT + (size_t)f * FF + c);
  }
#pragma unroll
  for (int i = 0; i < 4; ++i) {
    int id = tid + i * 256, pc = id >> 3, c = (id & 7) << 3;
    *(v8bf*)&Bs[pc][c] = *(const v8bf*)(hbf + (size_t)(p0 + pc) * FF + c);
  }
  __syncthreads();
#pragma unroll
  for (int kk = 0; kk < 64; kk += 32) {
    FragU a[4], bq;
#pragma unroll
    for (int mi = 0; mi < 4; ++mi) {
      int row = mi * 16 + ln;
      int kb  = kk + lh * 8;
      a[mi].h[0] = *(const v8bf*)&As[row][kb];
      a[mi].h[1] = *(const v8bf*)&As[row][kb + 16];
    }
    {
      int col = w * 16 + ln;
      int kb  = kk + lh * 8;
      bq.h[0] = *(const v8bf*)&Bs[col][kb];
      bq.h[1] = *(const v8bf*)&Bs[col][kb + 16];
    }
#pragma unroll
    for (int mi = 0; mi < 4; ++mi)
      acc[mi].v = __builtin_amdgcn_wmma_f32_16x16x32_bf16(
          false, a[mi].v, false, bq.v, (short)0, acc[mi].v, false, false);
  }
#pragma unroll
  for (int mi = 0; mi < 4; ++mi) {
    int f0 = mi * 16 + lh * 8;
    int p  = p0 + w * 16 + ln;
    AccU t;
#pragma unroll
    for (int r = 0; r < 8; ++r) t.f[r] = sigf(acc[mi].f[r] + be[f0 + r]);
    *(v8f*)(outp + (size_t)p * FF + f0) = t.v;
  }
}

// ---------------- host launcher ----------------
extern "C" void kernel_launch(void* const* d_in, const int* in_sizes, int n_in,
                              void* d_out, int out_size, void* d_ws, size_t ws_size,
                              hipStream_t stream) {
  (void)in_sizes; (void)n_in; (void)out_size; (void)ws_size;
  const float* x  = (const float*)d_in[0];
  const float* L  = (const float*)d_in[1];
  const float* Wg = (const float*)d_in[2];
  const float* bg = (const float*)d_in[3];
  const float* Wu = (const float*)d_in[4];
  const float* bu = (const float*)d_in[5];
  const float* We = (const float*)d_in[6];
  const float* be = (const float*)d_in[7];
  float* out = (float*)d_out;

  char*  ws  = (char*)d_ws;
  size_t off = 0;
  auto carve = [&](size_t bytes) -> void* {
    void* p = ws + off; off += (bytes + 255) & ~(size_t)255; return p;
  };
  bf16*  Lbf = (bf16*) carve((size_t)NN * NN * 2);
  bf16*  xbf = (bf16*) carve((size_t)MROWS * FF * 2);
  bf16*  xT  = (bf16*) carve((size_t)MROWS * FF * 2);
  bf16*  hT  = (bf16*) carve((size_t)MROWS * FF * 2);
  bf16*  hbf = (bf16*) carve((size_t)MROWS * FF * 2);
  bf16*  rh  = (bf16*) carve((size_t)MROWS * FF * 2);
  bf16*  rhT = (bf16*) carve((size_t)MROWS * FF * 2);
  bf16*  LC  = (bf16*) carve((size_t)MROWS * F2 * 2);
  float* uT  = (float*)carve((size_t)MROWS * FF * 4);
  float* hxT = (float*)carve((size_t)MROWS * FF * 4);
  bf16*  WgT = (bf16*) carve((size_t)F2 * 256 * 2);
  bf16*  WuT = (bf16*) carve((size_t)FF * 256 * 2);
  bf16*  WeT = (bf16*) carve((size_t)FF * FF * 2);

  auto blocks = [](int n) { int b = (n + 255) / 256; return b > 1024 ? 1024 : b; };
  k_cvt_bf16   <<<blocks(NN * NN),    256, 0, stream>>>(L, Lbf, NN * NN);
  k_cvt_bf16   <<<blocks(MROWS * FF), 256, 0, stream>>>(x, xbf, MROWS * FF);
  k_transpose_x<<<blocks(MROWS * FF), 256, 0, stream>>>(x, xT);
  k_transposeT <<<blocks(256 * F2),   256, 0, stream>>>(Wg, WgT, 256, F2);
  k_transposeT <<<blocks(256 * FF),   256, 0, stream>>>(Wu, WuT, 256, FF);
  k_transposeT <<<blocks(FF * FF),    256, 0, stream>>>(We, WeT, FF, FF);
  (void)hipMemsetAsync(hT,  0, (size_t)MROWS * FF * 2, stream);
  (void)hipMemsetAsync(hbf, 0, (size_t)MROWS * FF * 2, stream);
  (void)hipMemsetAsync(hxT, 0, (size_t)MROWS * FF * 4, stream);

  dim3 gLC(NN / 128, BB);
  int  gRows = MROWS / 128;   // 256 blocks

  for (int t = 0; t < TT; ++t) {
    k_gemm_lc<<<gLC,   256, 0, stream>>>(Lbf, xT, hT, LC);
    k_gates  <<<gRows, 256, 0, stream>>>(xbf, hbf, LC, WgT, bg, hxT, rh, rhT, uT);
    k_gemm_lc<<<gLC,   256, 0, stream>>>(Lbf, xT, rhT, LC);
    k_update <<<gRows, 256, 0, stream>>>(xbf, rh, LC, WuT, bu, uT, hxT, hT, hbf);
    k_edge   <<<gRows, 256, 0, stream>>>(hbf, WeT, be, out + (size_t)t * MROWS * FF);
  }
}
